// GAT_86947317940904
// MI455X (gfx1250) — matrix operand order repaired and checked by reading, not compiled, not run on hardware
//
#include <hip/hip_runtime.h>
#include <hip/hip_bf16.h>

#define NN 50000
#define NE 1600000
#define D1 128            // HEADS * HIDDEN = HEADS * OUT = 128 for both layers
#define NHEADS 8
#define HID 16
#define NEG_SLOPE 0.2f

typedef __attribute__((ext_vector_type(16))) _Float16 v16h;
typedef __attribute__((ext_vector_type(8)))  float    v8f;

// ---- order-preserving float<->uint encoding for atomic max ----
__device__ __forceinline__ unsigned ordEnc(float x) {
  unsigned u = __float_as_uint(x);
  return (u & 0x80000000u) ? ~u : (u | 0x80000000u);
}
__device__ __forceinline__ float ordDec(unsigned u) {
  return (u & 0x80000000u) ? __uint_as_float(u & 0x7FFFFFFFu)
                           : __uint_as_float(~u);
}
// ordEnc(-inf) == 0x007FFFFF
#define ORD_NEG_INF 0x007FFFFFu

__device__ __forceinline__ float lrelu(float v) {
  return v > 0.0f ? v : v * NEG_SLOPE;
}

// ======================= WMMA GEMM: C[NN,128] = A[NN,K] @ W[K,128] =======================
// block = 256 threads = 8 waves; wave w -> 16-column tile w; block -> 16-row tile blockIdx.x
__global__ __launch_bounds__(256)
void gat_gemm_wmma(const float* __restrict__ A, const float* __restrict__ W,
                   float* __restrict__ C, int K) {
  const int wave = threadIdx.x >> 5;
  const int lane = threadIdx.x & 31;
  const int hi   = lane >> 4;          // half-wave select
  const int l15  = lane & 15;
  const int row0 = blockIdx.x << 4;    // M tile base
  const int col  = (wave << 4) + l15;  // this lane's N column

  v8f acc = {};
  // A fragment per lane: halves 0..7  -> K = kk + hi*16 + i
  //                      halves 8..15 -> K = kk + hi*16 + 8 + i   (contiguous, matches B)
  const float* __restrict__ arow = A + (size_t)(row0 + l15) * (size_t)K + (size_t)(hi * 16);
  // B fragment per lane: halves i -> K = kk + hi*16 + i, column = col (contiguous k order, matches A)
  const float* __restrict__ wcol = W + (size_t)(hi * 16) * D1 + col;

  for (int kk = 0; kk < K; kk += 32) {
    v16h a, b;
#pragma unroll
    for (int i = 0; i < 8; ++i) {
      a[i]     = (_Float16)arow[kk + i];
      a[i + 8] = (_Float16)arow[kk + i + 8];
    }
#pragma unroll
    for (int i = 0; i < 16; ++i)
      b[i] = (_Float16)wcol[(size_t)(kk + i) * D1];
    acc = __builtin_amdgcn_wmma_f32_16x16x32_f16(
        /*neg_a=*/false, a, /*neg_b=*/false, b,
        /*c_mod=*/(short)0, acc, /*reuse_a=*/false, /*reuse_b=*/false);
  }
  // C/D layout: VGPR r holds (M = row0 + 8*hi + r, N = col)
#pragma unroll
  for (int r = 0; r < 8; ++r)
    C[(size_t)(row0 + (hi << 3) + r) * D1 + col] = acc[r];
}

// ======================= el/er: per (node, head) dot with attn vectors =======================
__global__ void gat_lr(const float* __restrict__ feat, const float* __restrict__ al,
                       const float* __restrict__ ar, float* __restrict__ el,
                       float* __restrict__ er, int total /* NN*NHEADS */) {
  int idx = blockIdx.x * blockDim.x + threadIdx.x;
  if (idx >= total) return;
  int h = idx & (NHEADS - 1);
  const float* f = feat + (size_t)(idx >> 3) * D1 + h * HID;
  float sl = 0.0f, sr = 0.0f;
#pragma unroll
  for (int d = 0; d < HID; ++d) {
    float x = f[d];
    sl += x * al[h * HID + d];
    sr += x * ar[h * HID + d];
  }
  el[idx] = sl;
  er[idx] = sr;
}

// ======================= init kernels =======================
__global__ void gat_init_soft(unsigned* __restrict__ mx, float* __restrict__ dn, int total) {
  int idx = blockIdx.x * blockDim.x + threadIdx.x;
  if (idx >= total) return;
  mx[idx] = ORD_NEG_INF;
  dn[idx] = 0.0f;
}
__global__ void gat_zero(float* __restrict__ p, int total) {
  int idx = blockIdx.x * blockDim.x + threadIdx.x;
  if (idx < total) p[idx] = 0.0f;
}

// ======================= edge pass 1: segment max =======================
__global__ void gat_edge_max(const int* __restrict__ src, const int* __restrict__ dst,
                             const float* __restrict__ el, const float* __restrict__ er,
                             unsigned* __restrict__ mx, int total /* NE*NHEADS */) {
  int idx = blockIdx.x * blockDim.x + threadIdx.x;
  if (idx >= total) return;
  int e = idx >> 3, h = idx & (NHEADS - 1);
  int s = src[e], t = dst[e];
  float v = lrelu(el[s * NHEADS + h] + er[t * NHEADS + h]);
  atomicMax(&mx[t * NHEADS + h], ordEnc(v));
}

// ======================= edge pass 2: segment sum of exp =======================
__global__ void gat_edge_den(const int* __restrict__ src, const int* __restrict__ dst,
                             const float* __restrict__ el, const float* __restrict__ er,
                             const unsigned* __restrict__ mx, float* __restrict__ dn,
                             int total /* NE*NHEADS */) {
  int idx = blockIdx.x * blockDim.x + threadIdx.x;
  if (idx >= total) return;
  int e = idx >> 3, h = idx & (NHEADS - 1);
  int s = src[e], t = dst[e];
  float v = lrelu(el[s * NHEADS + h] + er[t * NHEADS + h]);
  float ex = expf(v - ordDec(mx[t * NHEADS + h]));
  atomicAdd(&dn[t * NHEADS + h], ex);
}

// ======================= edge pass 3: alpha-weighted scatter of feat[src] =======================
// one thread per (edge, head, d) -> coalesced feat reads / agg atomics
__global__ void gat_edge_agg(const int* __restrict__ src, const int* __restrict__ dst,
                             const float* __restrict__ el, const float* __restrict__ er,
                             const unsigned* __restrict__ mx, const float* __restrict__ dn,
                             const float* __restrict__ feat, float* __restrict__ agg,
                             int total /* NE*128 */) {
  int idx = blockIdx.x * blockDim.x + threadIdx.x;
  if (idx >= total) return;
  int d = idx & (HID - 1);
  int h = (idx >> 4) & (NHEADS - 1);
  int e = idx >> 7;
  int s = src[e], t = dst[e];
  float v = lrelu(el[s * NHEADS + h] + er[t * NHEADS + h]);
  float alpha = expf(v - ordDec(mx[t * NHEADS + h])) / dn[t * NHEADS + h];
  int c = h * HID + d;
  atomicAdd(&agg[(size_t)t * D1 + c], alpha * feat[(size_t)s * D1 + c]);
}

// ======================= epilogue: h = elu(agg + bias), in place =======================
__global__ void gat_elu_bias(float* __restrict__ agg, const float* __restrict__ b,
                             int total /* NN*128 */) {
  int idx = blockIdx.x * blockDim.x + threadIdx.x;
  if (idx >= total) return;
  float v = agg[idx] + b[idx & (D1 - 1)];
  agg[idx] = v > 0.0f ? v : (expf(v) - 1.0f);
}

// ======================= final: mean over heads =======================
__global__ void gat_final(const float* __restrict__ agg2, const float* __restrict__ b2,
                          float* __restrict__ out, int total /* NN*16 */) {
  int idx = blockIdx.x * blockDim.x + threadIdx.x;
  if (idx >= total) return;
  int j = idx & (HID - 1);
  int n = idx >> 4;
  float sacc = 0.0f;
#pragma unroll
  for (int h = 0; h < NHEADS; ++h)
    sacc += agg2[(size_t)n * D1 + h * HID + j] + b2[h * HID + j];
  out[idx] = sacc * (1.0f / NHEADS);
}

extern "C" void kernel_launch(void* const* d_in, const int* in_sizes, int n_in,
                              void* d_out, int out_size, void* d_ws, size_t ws_size,
                              hipStream_t stream) {
  const float* node_feat = (const float*)d_in[0];
  const int*   src       = (const int*)d_in[1];
  const int*   dst       = (const int*)d_in[2];
  const float* W1        = (const float*)d_in[3];
  const float* al1       = (const float*)d_in[4];
  const float* ar1       = (const float*)d_in[5];
  const float* b1        = (const float*)d_in[6];
  const float* W2        = (const float*)d_in[7];
  const float* al2       = (const float*)d_in[8];
  const float* ar2       = (const float*)d_in[9];
  const float* b2        = (const float*)d_in[10];
  float* out = (float*)d_out;

  // ---- workspace carve-out (~83 MB) ----
  float* feat = (float*)d_ws;                         // NN*128  (feat1, later reused as feat2)
  float* agg1 = feat + (size_t)NN * D1;               // NN*128  (becomes h after elu)
  float* agg2 = agg1 + (size_t)NN * D1;               // NN*128
  float* el   = agg2 + (size_t)NN * D1;               // NN*8
  float* er   = el + (size_t)NN * NHEADS;             // NN*8
  unsigned* mx = (unsigned*)(er + (size_t)NN * NHEADS); // NN*8
  float* dn   = (float*)(mx + (size_t)NN * NHEADS);   // NN*8

  const int TPB = 256;
  const int nh_total   = NN * NHEADS;        // 400,000
  const int nf_total   = NN * D1;            // 6,400,000
  const int eh_total   = NE * NHEADS;        // 12,800,000
  const int ed_total   = NE * D1;            // 204,800,000
  const int out_total  = NN * HID;           // 800,000
  const int gemm_grid  = NN / 16;            // 3125 (exact)

  auto cdiv = [](int a, int b) { return (a + b - 1) / b; };

  // ---- init (every call: graph-replay deterministic) ----
  gat_init_soft<<<cdiv(nh_total, TPB), TPB, 0, stream>>>(mx, dn, nh_total);
  gat_zero<<<cdiv(2 * nf_total, TPB), TPB, 0, stream>>>(agg1, 2 * nf_total); // agg1 + agg2

  // ================= layer 1 =================
  gat_gemm_wmma<<<gemm_grid, TPB, 0, stream>>>(node_feat, W1, feat, 256);
  gat_lr<<<cdiv(nh_total, TPB), TPB, 0, stream>>>(feat, al1, ar1, el, er, nh_total);
  gat_edge_max<<<cdiv(eh_total, TPB), TPB, 0, stream>>>(src, dst, el, er, mx, eh_total);
  gat_edge_den<<<cdiv(eh_total, TPB), TPB, 0, stream>>>(src, dst, el, er, mx, dn, eh_total);
  gat_edge_agg<<<cdiv(ed_total, TPB), TPB, 0, stream>>>(src, dst, el, er, mx, dn, feat, agg1, ed_total);
  gat_elu_bias<<<cdiv(nf_total, TPB), TPB, 0, stream>>>(agg1, b1, nf_total); // agg1 := h

  // ================= layer 2 =================
  gat_gemm_wmma<<<gemm_grid, TPB, 0, stream>>>(agg1, W2, feat, 128);          // feat := feat2
  gat_lr<<<cdiv(nh_total, TPB), TPB, 0, stream>>>(feat, al2, ar2, el, er, nh_total);
  gat_init_soft<<<cdiv(nh_total, TPB), TPB, 0, stream>>>(mx, dn, nh_total);   // reset softmax state
  gat_edge_max<<<cdiv(eh_total, TPB), TPB, 0, stream>>>(src, dst, el, er, mx, eh_total);
  gat_edge_den<<<cdiv(eh_total, TPB), TPB, 0, stream>>>(src, dst, el, er, mx, dn, eh_total);
  gat_edge_agg<<<cdiv(ed_total, TPB), TPB, 0, stream>>>(src, dst, el, er, mx, dn, feat, agg2, ed_total);

  // ================= mean over heads =================
  gat_final<<<cdiv(out_total, TPB), TPB, 0, stream>>>(agg2, b2, out, out_total);
}
